// SelfAttentionHead_5162550690687
// MI455X (gfx1250) — compile-verified, matrix operand order checked by reading, not run on hardware
//
#include <hip/hip_runtime.h>
#include <hip/hip_bf16.h>

typedef __attribute__((ext_vector_type(16))) _Float16 v16h;
typedef __attribute__((ext_vector_type(8)))  float    v8f;
typedef __attribute__((ext_vector_type(4)))  float    v4f;

#define BQ 4
#define TQ 4096
#define CQ 1024
#define HQ 64

__device__ __forceinline__ v8f wmma_f16(v16h a, v16h b, v8f c) {
    // D = A(16x32 f16) * B(32x16 f16) + C(16x16 f32)
    return __builtin_amdgcn_wmma_f32_16x16x32_f16(
        /*neg_a=*/false, a, /*neg_b=*/false, b,
        /*c_mod=*/(short)0, c, /*reuse_a=*/false, /*reuse_b=*/false);
}

// ---------------------------------------------------------------------------
// Kernel 0: transpose+convert weights.  Wt[mat][n][k] = W_mat[k][n] as f16.
// ---------------------------------------------------------------------------
__global__ void prep_w_kernel(const float* __restrict__ Wq,
                              const float* __restrict__ Wk,
                              const float* __restrict__ Wv,
                              _Float16* __restrict__ Wt) {
    int idx = blockIdx.x * blockDim.x + threadIdx.x;   // 3*64*1024
    if (idx >= 3 * HQ * CQ) return;
    int mat = idx >> 16;          // / (64*1024)
    int rem = idx & 65535;
    int n   = rem >> 10;          // / 1024
    int k   = rem & 1023;
    const float* W = (mat == 0) ? Wq : ((mat == 1) ? Wk : Wv);
    Wt[idx] = (_Float16)W[k * HQ + n];
}

// ---------------------------------------------------------------------------
// Kernel 1: q/k/v projections via WMMA.  One wave per 16-row tile of x.
//   qf[t][h] = (x @ Wq)[t][h] * H^-0.5   (f16, row-major)
//   kf[t][h] = (x @ Wk)[t][h]            (f16, row-major)
//   vt[b][h][t] = (x @ Wv)[t][h]         (f16, dim-major / transposed)
// ---------------------------------------------------------------------------
__global__ void proj_kernel(const float* __restrict__ x,
                            const _Float16* __restrict__ Wt,
                            _Float16* __restrict__ qf,
                            _Float16* __restrict__ kf,
                            _Float16* __restrict__ vt) {
    const int lane = threadIdx.x & 31;
    const int wave = blockIdx.x * (blockDim.x >> 5) + (threadIdx.x >> 5);
    const int m0   = wave * 16;              // row tile in [0, B*T)
    const int half = lane >> 4;              // 0: K 0..15, 1: K 16..31
    const int lm   = lane & 15;

    v8f acc[3][4];
    #pragma unroll
    for (int m = 0; m < 3; ++m)
        #pragma unroll
        for (int nb = 0; nb < 4; ++nb)
            acc[m][nb] = (v8f){0.f,0.f,0.f,0.f,0.f,0.f,0.f,0.f};

    const float* xrow = x + (size_t)(m0 + lm) * CQ + half * 16;

    for (int k0 = 0; k0 < CQ; k0 += 32) {
        // A operand: 16 rows of x, this lane: row (m0+lm), dims k0+half*16..+15
        v16h a;
        const v4f* xp = (const v4f*)(xrow + k0);
        #pragma unroll
        for (int i = 0; i < 4; ++i) {
            v4f f = xp[i];
            #pragma unroll
            for (int j = 0; j < 4; ++j) a[4 * i + j] = (_Float16)f[j];
        }
        #pragma unroll
        for (int m = 0; m < 3; ++m) {
            #pragma unroll
            for (int nb = 0; nb < 4; ++nb) {
                const v16h* bp = (const v16h*)(Wt + (size_t)m * (HQ * CQ)
                                               + (nb * 16 + lm) * CQ
                                               + k0 + half * 16);
                acc[m][nb] = wmma_f16(a, *bp, acc[m][nb]);
            }
        }
    }

    const float qscale = 0.125f;   // H^-0.5 folded into q
    #pragma unroll
    for (int nb = 0; nb < 4; ++nb) {
        #pragma unroll
        for (int r = 0; r < 8; ++r) {
            int mrow = m0 + r + half * 8;
            int n    = nb * 16 + lm;
            qf[(size_t)mrow * HQ + n] = (_Float16)(acc[0][nb][r] * qscale);
            kf[(size_t)mrow * HQ + n] = (_Float16)(acc[1][nb][r]);
            int b = mrow >> 12;            // / 4096
            int t = mrow & 4095;
            vt[((size_t)b * HQ + n) * TQ + t] = (_Float16)(acc[2][nb][r]);
        }
    }
}

// ---------------------------------------------------------------------------
// Kernel 2: causal flash attention (transposed layout).
// One wave handles a 16-query tile; each lane owns one query column.
//   S^T = K * Q^T  (per 32-key block, two 16x16 f32 accumulators)
//   online softmax over keys (scalar stats per lane + one xor-16 shuffle)
//   O^T += V^T * P^T  (P^T built in registers via one half-wave swap)
// ---------------------------------------------------------------------------
__global__ void attn_kernel(const _Float16* __restrict__ qf,
                            const _Float16* __restrict__ kf,
                            const _Float16* __restrict__ vt,
                            float* __restrict__ out) {
    const int lane = threadIdx.x & 31;
    const int wave = blockIdx.x * (blockDim.x >> 5) + (threadIdx.x >> 5);
    const int b    = wave >> 8;               // 256 tiles per batch
    const int i0   = (wave & 255) * 16;       // query tile start
    const int half = lane >> 4;
    const int lm   = lane & 15;
    const int q_g  = i0 + lm;                 // this lane's query index

    // Q^T operands (B-matrix): lane holds query q_g, 16 dims per half.
    const _Float16* qrow = qf + (size_t)(b * TQ + i0 + lm) * HQ + half * 16;
    const v16h bq0 = *(const v16h*)(qrow);        // dims  0..31
    const v16h bq1 = *(const v16h*)(qrow + 32);   // dims 32..63

    v8f ot[4];
    #pragma unroll
    for (int nb = 0; nb < 4; ++nb)
        ot[nb] = (v8f){0.f,0.f,0.f,0.f,0.f,0.f,0.f,0.f};
    float m_run = -1e30f, l_run = 0.f;

    for (int j0 = 0; j0 < i0 + 16; j0 += 32) {
        v8f st0 = (v8f){0.f,0.f,0.f,0.f,0.f,0.f,0.f,0.f};
        v8f st1 = st0;
        const _Float16* krow0 = kf + (size_t)(b * TQ + j0 + lm) * HQ + half * 16;
        const _Float16* krow1 = krow0 + 16 * HQ;
        st0 = wmma_f16(*(const v16h*)(krow0),      bq0, st0);
        st0 = wmma_f16(*(const v16h*)(krow0 + 32), bq1, st0);
        st1 = wmma_f16(*(const v16h*)(krow1),      bq0, st1);
        st1 = wmma_f16(*(const v16h*)(krow1 + 32), bq1, st1);

        // causal mask + block max (st[r] = score(key j0[+16]+r+8*half, q_g))
        float pmax = -1e30f;
        #pragma unroll
        for (int r = 0; r < 8; ++r) {
            int key0 = j0 + r + half * 8;
            float s0 = (key0      <= q_g) ? st0[r] : -1e30f;
            float s1 = (key0 + 16 <= q_g) ? st1[r] : -1e30f;
            st0[r] = s0; st1[r] = s1;
            pmax = fmaxf(pmax, fmaxf(s0, s1));
        }
        pmax = fmaxf(pmax, __shfl_xor(pmax, 16));
        float m_new = fmaxf(m_run, pmax);
        float alpha = __expf(m_run - m_new);

        float p0[8], p1[8], psum = 0.f;
        #pragma unroll
        for (int r = 0; r < 8; ++r) {
            p0[r] = __expf(st0[r] - m_new);
            p1[r] = __expf(st1[r] - m_new);
            psum += p0[r] + p1[r];
        }
        psum += __shfl_xor(psum, 16);
        l_run = l_run * alpha + psum;
        m_run = m_new;

        #pragma unroll
        for (int nb = 0; nb < 4; ++nb)
            #pragma unroll
            for (int r = 0; r < 8; ++r) ot[nb][r] *= alpha;

        // Build P^T B-operand: lane needs keys 16*half+0..15 for its query.
        v16h bp;
        #pragma unroll
        for (int r = 0; r < 8; ++r) {
            float t0 = __shfl_xor(p0[r], 16);  // lower<-keys 8..15, upper<-keys 0..7
            float t1 = __shfl_xor(p1[r], 16);  // upper<-keys 16..23
            float first  = half ? t1    : p0[r];   // key 16*half + r
            float second = half ? p1[r] : t0;      // key 16*half + 8 + r
            bp[r]     = (_Float16)first;
            bp[8 + r] = (_Float16)second;
        }

        // O^T += V^T * P^T : A-operand lane holds dim nb*16+lm, keys j0+half*16..+15
        #pragma unroll
        for (int nb = 0; nb < 4; ++nb) {
            const v16h* vp = (const v16h*)(vt + ((size_t)b * HQ + nb * 16 + lm) * TQ
                                           + j0 + half * 16);
            ot[nb] = wmma_f16(*vp, bp, ot[nb]);
        }
    }

    float inv_l = 1.0f / l_run;
    #pragma unroll
    for (int nb = 0; nb < 4; ++nb) {
        #pragma unroll
        for (int r = 0; r < 8; ++r) {
            int dim = nb * 16 + r + half * 8;
            out[(size_t)(b * TQ + i0 + lm) * HQ + dim] = ot[nb][r] * inv_l;
        }
    }
}

// ---------------------------------------------------------------------------
extern "C" void kernel_launch(void* const* d_in, const int* in_sizes, int n_in,
                              void* d_out, int out_size, void* d_ws, size_t ws_size,
                              hipStream_t stream) {
    const float* x  = (const float*)d_in[0];
    const float* Wq = (const float*)d_in[1];
    const float* Wk = (const float*)d_in[2];
    const float* Wv = (const float*)d_in[3];
    float* out = (float*)d_out;

    char* ws = (char*)d_ws;
    _Float16* Wt = (_Float16*)(ws);                               // 3*64*1024*2 = 0x60000 B
    _Float16* qf = (_Float16*)(ws + 0x60000);                     // 16384*64*2  = 0x200000 B
    _Float16* kf = (_Float16*)(ws + 0x60000 + 0x200000);          // 0x200000 B
    _Float16* vt = (_Float16*)(ws + 0x60000 + 0x400000);          // 0x200000 B

    // 3*64*1024 = 196608 elements
    prep_w_kernel<<<768, 256, 0, stream>>>(Wq, Wk, Wv, Wt);
    // 16384/16 = 1024 waves, 4 waves per 128-thread block
    proj_kernel<<<256, 128, 0, stream>>>(x, Wt, qf, kf, vt);
    // 1024 query tiles, 4 waves per block
    attn_kernel<<<256, 128, 0, stream>>>(qf, kf, vt, out);
}
